// BuckeFillingRNNTModel_32899449487467
// MI455X (gfx1250) — matrix operand order repaired and checked by reading, not compiled, not run on hardware
//
#include <hip/hip_runtime.h>
#include <hip/hip_bf16.h>

// ---------------------------------------------------------------------------
// Types for CDNA5 WMMA (wave32): D(f32 16x16) = A(bf16 16x32) * B(bf16 32x16) + C
// ---------------------------------------------------------------------------
typedef __attribute__((ext_vector_type(16))) __bf16 v16bf;
typedef __attribute__((ext_vector_type(8)))  __bf16 v8bf;
typedef __attribute__((ext_vector_type(8)))  float  v8f;
typedef __attribute__((ext_vector_type(4)))  unsigned v4u;
typedef __attribute__((ext_vector_type(8)))  int      v8i;
typedef __attribute__((ext_vector_type(4)))  int      v4i;

static __device__ __forceinline__ v8f wmma_bf16(v16bf a, v16bf b, v8f c) {
  return __builtin_amdgcn_wmma_f32_16x16x32_bf16(false, a, false, b, (short)0, c, false, false);
}

static __device__ __forceinline__ v16bf join16(v8bf lo, v8bf hi) {
  v16bf r;
#pragma unroll
  for (int i = 0; i < 8; ++i) { r[i] = lo[i]; r[i + 8] = hi[i]; }
  return r;
}

// A fragment (row-major A[M,K]):
//  lanes 0-15 : M=lane, elems 0..15 = K k0+0..15 contiguous
//  lanes 16-31: M=lane-16, elems 0..15 = K k0+16..31 contiguous
static __device__ __forceinline__ v16bf load_afrag(const __bf16* __restrict__ A, int lda,
                                                   int row0, int k0, int lane) {
  const int m    = row0 + (lane & 15);
  const int half = lane >> 4;
  const __bf16* p = A + (long)m * lda + k0 + 16 * half;
  return join16(*(const v8bf*)p, *(const v8bf*)(p + 8));
}

// B fragment: weights W[N,K] row-major == B^T.
//  lanes 0-15 : N=lane, K=k0..k0+15 contiguous; lanes 16-31: N=lane-16, K=k0+16..31
static __device__ __forceinline__ v16bf load_bfrag(const __bf16* __restrict__ Bw, int ldb,
                                                   int n0, int k0, int lane) {
  const int n    = n0 + (lane & 15);
  const int half = lane >> 4;
  const __bf16* p = Bw + (long)n * ldb + k0 + 16 * half;
  return join16(*(const v8bf*)p, *(const v8bf*)(p + 8));
}

// CDNA5 async global->LDS copy (ASYNCcnt): 16 bytes per lane.
static __device__ __forceinline__ void async_copy_b128(unsigned lds_off, const void* gptr) {
  asm volatile("global_load_async_to_lds_b128 %0, %1, off"
               :: "v"(lds_off), "v"(gptr) : "memory");
}
static __device__ __forceinline__ void async_wait0() {
  asm volatile("s_wait_asynccnt 0x0" ::: "memory");
}

// ---------------------------------------------------------------------------
// Problem constants
// ---------------------------------------------------------------------------
#define BATCH 16
#define TLEN  2000
#define FIN   240
#define KX    256
#define HDIM  512
#define GDIM  1024
#define MROWS 32000
#define NWG   16
#define LPAD  520            // 512 + 8 bf16 pad -> 1040B row stride (bank-conflict-free)

// ---------------------------------------------------------------------------
// CDNA5 Tensor Data Mover: 2D tile (rows x 512 bf16) global->LDS with HW row
// padding: pad_interval=7 (256 dwords = one 1024B row), pad_amount=3 (4 dwords
// = 16B) -> LDS row stride 1040B == LPAD*2. Groups 2/3 zero (tile_dim2/3 = 0).
// This toolchain uses the 6-arg builtin: (v4u, v8i, v4i, v4i, v8i, cpol).
// ---------------------------------------------------------------------------
static __device__ __forceinline__ void tdm_load_2d_rows512(unsigned lds_addr,
                                                           const void* gptr,
                                                           unsigned rows) {
  unsigned long long ga = (unsigned long long)(uintptr_t)gptr;
  v4u g0;
  g0[0] = 1u;                                               // count=1 (valid user D#)
  g0[1] = lds_addr;                                         // LDS byte address
  g0[2] = (unsigned)(ga & 0xFFFFFFFFu);                     // global_addr[31:0]
  g0[3] = (unsigned)((ga >> 32) & 0x01FFFFFFu) | (2u << 30);// global_addr[56:32] | type=2
  v8i g1;
  g1[0] = (int)((1u << 16) | (1u << 20) | (7u << 22) | (3u << 25)); // data_size=2B, pad en/intvl/amt
  g1[1] = (int)(512u << 16);                                // tensor_dim0[15:0] -> bits 63:48
  g1[2] = (int)((512u >> 16) | (1024u << 16));              // tensor_dim0 hi | tensor_dim1 lo
  g1[3] = (int)((1024u >> 16) | (512u << 16));              // tensor_dim1 hi | tile_dim0=512
  g1[4] = (int)rows;                                        // tile_dim1=rows, tile_dim2=0
  g1[5] = 512;                                              // tensor_dim0_stride[31:0]
  g1[6] = 0;
  g1[7] = 0;
  v4i z4 = {0, 0, 0, 0};
  v8i z8 = {0, 0, 0, 0, 0, 0, 0, 0};
  __builtin_amdgcn_tensor_load_to_lds(g0, g1, z4, z4, z8, 0);
}

// ---------------------------------------------------------------------------
// Workspace layout (bytes, all 256-aligned)
// ---------------------------------------------------------------------------
static constexpr long OFF_XB   = 0;                          // bf16 [32000][256]
static constexpr long OFF_WX   = OFF_XB   + 16384000L;       // f32  [32000][1024]
static constexpr long OFF_HBUF = OFF_WX   + 131072000L;      // bf16 [32000][512]
static constexpr long OFF_W0B  = OFF_HBUF + 32768000L;       // bf16 [1024][256]
static constexpr long OFF_W1B  = OFF_W0B  + 524288L;         // bf16 [1024][512]
static constexpr long OFF_WDB  = OFF_W1B  + 1048576L;        // bf16 [1024][512]
static constexpr long OFF_U0B  = OFF_WDB  + 1048576L;        // bf16 [1024][512]
static constexpr long OFF_U1B  = OFF_U0B  + 1048576L;        // bf16 [1024][512]
static constexpr long OFF_BN   = OFF_U1B  + 1048576L;        // f32 sum/sqs/scale/shift 4x1024
static constexpr long OFF_LN   = OFF_BN   + 16384L;          // f32 [2 layers][2 buf][32]
static constexpr long OFF_HG   = OFF_LN   + 512L;            // bf16 [2][16][512]
static constexpr long OFF_BAR  = OFF_HG   + 32768L;          // int barriers
static constexpr long WS_NEED  = OFF_BAR  + 256L;
static constexpr long ZERO_U32 = (16384L + 512L + 32768L + 256L) / 4;

// ---------------------------------------------------------------------------
// Utility kernels
// ---------------------------------------------------------------------------
__global__ void zero_kernel(unsigned* __restrict__ p, long n) {
  for (long i = blockIdx.x * (long)blockDim.x + threadIdx.x; i < n;
       i += (long)gridDim.x * blockDim.x)
    p[i] = 0u;
}

__global__ void convert_pad_kernel(const float* __restrict__ src, int sR, int sC,
                                   __bf16* __restrict__ dst, int dR, int dC) {
  long n = (long)dR * dC;
  for (long i = blockIdx.x * (long)blockDim.x + threadIdx.x; i < n;
       i += (long)gridDim.x * blockDim.x) {
    int r = (int)(i / dC), c = (int)(i % dC);
    float v = (r < sR && c < sC) ? src[(long)r * sC + c] : 0.f;
    dst[i] = (__bf16)v;
  }
}

__global__ void bn_finalize_kernel(float* __restrict__ bnsum, float* __restrict__ bnsqs,
                                   const float* __restrict__ g, const float* __restrict__ b,
                                   float* __restrict__ scale, float* __restrict__ shift) {
  int i = blockIdx.x * blockDim.x + threadIdx.x;
  if (i < GDIM) {
    const float inv = 1.f / (float)MROWS;
    float mu  = bnsum[i] * inv;
    float var = bnsqs[i] * inv - mu * mu;
    float sc  = g[i] * rsqrtf(var + 1e-5f);
    scale[i] = sc;
    shift[i] = b[i] - mu * sc;
    bnsum[i] = 0.f;
    bnsqs[i] = 0.f;
  }
}

// ---------------------------------------------------------------------------
// wx GEMM: C[M,1024] = A[M,K](bf16) * W[1024,K]^T + per-column sum/sumsq.
// 256 threads = 8 waves in 2(M) x 4(N); per-wave 32x64 -> 8 WMMA per k-chunk.
// ---------------------------------------------------------------------------
__global__ __launch_bounds__(256)
void wx_gemm_kernel(const __bf16* __restrict__ A, int lda, int K,
                    const __bf16* __restrict__ Bw,
                    float* __restrict__ C,
                    float* __restrict__ bnsum, float* __restrict__ bnsqs) {
  const int tid  = threadIdx.x;
  const int lane = tid & 31;
  const int wave = tid >> 5;
  const int mw = wave >> 2, nw = wave & 3;
  const int m0 = blockIdx.y * 64 + mw * 32;
  const int n0 = blockIdx.x * 256 + nw * 64;
  const int half = lane >> 4, nl = lane & 15;

  v8f d[2][4] = {};
  for (int k0 = 0; k0 < K; k0 += 32) {
    if (k0 + 32 < K) {
      __builtin_prefetch(A + (long)(m0 + nl) * lda + k0 + 32, 0, 1);
    }
    v16bf a0 = load_afrag(A, lda, m0,      k0, lane);
    v16bf a1 = load_afrag(A, lda, m0 + 16, k0, lane);
#pragma unroll
    for (int ni = 0; ni < 4; ++ni) {
      v16bf b = load_bfrag(Bw, K, n0 + ni * 16, k0, lane);
      d[0][ni] = wmma_bf16(a0, b, d[0][ni]);
      d[1][ni] = wmma_bf16(a1, b, d[1][ni]);
    }
  }

  __shared__ float cs[256], cq[256];
  cs[tid] = 0.f; cq[tid] = 0.f;
  __syncthreads();

  float s1[4] = {0.f, 0.f, 0.f, 0.f}, s2[4] = {0.f, 0.f, 0.f, 0.f};
#pragma unroll
  for (int mi = 0; mi < 2; ++mi)
#pragma unroll
    for (int ni = 0; ni < 4; ++ni)
#pragma unroll
      for (int r = 0; r < 8; ++r) {
        float v = d[mi][ni][r];
        int row = m0 + mi * 16 + r + 8 * half;
        int col = n0 + ni * 16 + nl;
        C[(long)row * GDIM + col] = v;
        s1[ni] += v; s2[ni] += v * v;
      }
#pragma unroll
  for (int ni = 0; ni < 4; ++ni) {
    int lc = nw * 64 + ni * 16 + nl;
    atomicAdd(&cs[lc], s1[ni]);
    atomicAdd(&cq[lc], s2[ni]);
  }
  __syncthreads();
  atomicAdd(&bnsum[blockIdx.x * 256 + tid], cs[tid]);
  atomicAdd(&bnsqs[blockIdx.x * 256 + tid], cq[tid]);
}

// ---------------------------------------------------------------------------
// Head GEMM: out[M,1000] = h[M,512] * Wd[1024,512]^T + bd  (cols >= 1000 masked)
// ---------------------------------------------------------------------------
__global__ __launch_bounds__(256)
void head_gemm_kernel(const __bf16* __restrict__ A, const __bf16* __restrict__ Bw,
                      const float* __restrict__ bias, float* __restrict__ out) {
  const int tid  = threadIdx.x;
  const int lane = tid & 31;
  const int wave = tid >> 5;
  const int mw = wave >> 2, nw = wave & 3;
  const int m0 = blockIdx.y * 64 + mw * 32;
  const int n0 = blockIdx.x * 256 + nw * 64;
  const int half = lane >> 4, nl = lane & 15;

  v8f d[2][4] = {};
  for (int k0 = 0; k0 < HDIM; k0 += 32) {
    if (k0 + 32 < HDIM) __builtin_prefetch(A + (long)(m0 + nl) * HDIM + k0 + 32, 0, 1);
    v16bf a0 = load_afrag(A, HDIM, m0,      k0, lane);
    v16bf a1 = load_afrag(A, HDIM, m0 + 16, k0, lane);
#pragma unroll
    for (int ni = 0; ni < 4; ++ni) {
      v16bf b = load_bfrag(Bw, HDIM, n0 + ni * 16, k0, lane);
      d[0][ni] = wmma_bf16(a0, b, d[0][ni]);
      d[1][ni] = wmma_bf16(a1, b, d[1][ni]);
    }
  }
#pragma unroll
  for (int mi = 0; mi < 2; ++mi)
#pragma unroll
    for (int ni = 0; ni < 4; ++ni) {
      int col = n0 + ni * 16 + nl;
      if (col < 1000) {
        float bv = bias[col];
#pragma unroll
        for (int r = 0; r < 8; ++r) {
          int row = m0 + mi * 16 + r + 8 * half;
          out[(long)row * 1000 + col] = d[mi][ni][r] + bv;
        }
      }
    }
}

// ---------------------------------------------------------------------------
// Global barrier across the 16 cooperative WGs (monotonic counter) with the
// CDNA5 cluster barrier (NOP when not launched as a cluster).
// ---------------------------------------------------------------------------
static __device__ __forceinline__ void gbarrier(int* __restrict__ bar, int idx, int tid) {
  __threadfence();
  __syncthreads();
  if (tid == 0) {
    __builtin_amdgcn_s_cluster_barrier();
    atomicAdd(bar, 1);
    const int target = (idx + 1) * NWG;
    while (__hip_atomic_load(bar, __ATOMIC_RELAXED, __HIP_MEMORY_SCOPE_AGENT) < target) {
      __builtin_amdgcn_s_sleep(2);
    }
  }
  __syncthreads();
  __threadfence();
}

// ---------------------------------------------------------------------------
// SLi-GRU recurrence: 16 cooperative WGs x 128 threads (4 waves).
// WG g owns h cols [32g,32g+32): gate tiles {2g,2g+1} (a) and {32+2g,33+2g} (z).
// U slice staged to LDS ONCE via TDM (tensor_load_to_lds, HW row padding);
// h panel staged to LDS every step via async-to-LDS. WMMA operands from LDS.
// ---------------------------------------------------------------------------
__global__ __launch_bounds__(128, 1)
void sligru_recur_kernel(const __bf16* __restrict__ Ub,       // [1024][512] bf16
                         const float*  __restrict__ wx,       // [32000][1024] raw
                         const float*  __restrict__ bnscale,
                         const float*  __restrict__ bnshift,
                         const float*  __restrict__ ln_g,
                         const float*  __restrict__ ln_b,
                         __bf16* __restrict__ hbuf,            // [32000][512] bf16 out
                         __bf16* __restrict__ h_g,             // [16][512] bf16 shared state
                         float*  __restrict__ lnstat,          // [2][32]
                         int*    __restrict__ bar) {
  const int g    = blockIdx.x;
  const int tid  = threadIdx.x;
  const int lane = tid & 31;
  const int w    = tid >> 5;
  const int half = lane >> 4;
  const int nl   = lane & 15;
  const int tn   = (w < 2) ? (2 * g + w) : (32 + 2 * g + (w - 2));
  const int col  = tn * 16 + nl;
  const int hcol = 32 * g + ((w < 2) ? w : (w - 2)) * 16 + nl;

  __shared__ __align__(16) __bf16 UbS[4][16][LPAD];   // 66,560 B
  __shared__ __align__(16) __bf16 hS[16][LPAD];       // 16,640 B
  __shared__ float zS[16][32];
  __shared__ float rS1[16], rS2[16];
  __shared__ float stS1[16], stS2[16];

  const unsigned ub_base = (unsigned)(uintptr_t)&UbS[0][0][0];
  const unsigned hs_base = (unsigned)(uintptr_t)&hS[0][0];

  // ---- Stage U slice into LDS once via TDM (two contiguous 32x512 tiles) ----
  if (w == 0) {
    tdm_load_2d_rows512(ub_base, Ub + (long)(32 * g) * HDIM, 32);            // a-rows
  } else if (w == 1) {
    tdm_load_2d_rows512(ub_base + 2u * 16u * LPAD * 2u,
                        Ub + (long)(512 + 32 * g) * HDIM, 32);               // z-rows
  }
  __builtin_amdgcn_s_wait_tensorcnt(0);
  __syncthreads();

  const float lng = ln_g[col], lnb = ln_b[col];
  const float bsc = bnscale[col], bsh = bnshift[col];
  float hreg[8];
#pragma unroll
  for (int r = 0; r < 8; ++r) hreg[r] = 0.f;

  int barIdx = 0;
  for (int t = 0; t < TLEN; ++t) {
    if (g == 0 && tid < 32) lnstat[((t + 1) & 1) * 32 + tid] = 0.f;
    if (tid < 16) rS1[tid] = 0.f; else if (tid < 32) rS2[tid - 16] = 0.f;

    // ---- Stage h panel into LDS: 16 rows x 64 16B-chunks (async-to-LDS) ----
    for (int c = tid; c < 1024; c += 128) {
      int m = c >> 6, q = c & 63;
      async_copy_b128(hs_base + (unsigned)((m * LPAD) * 2 + q * 16),
                      (const char*)h_g + ((long)m * HDIM) * 2 + q * 16);
    }
    async_wait0();
    __syncthreads();

    // u-tile = h[16x512] x U^T tile -- 16 chained WMMAs, operands from LDS
    v8f acc = {};
#pragma unroll 4
    for (int k0 = 0; k0 < HDIM; k0 += 32) {
      const __bf16* ap = &hS[nl][k0 + 16 * half];
      v16bf a = join16(*(const v8bf*)ap, *(const v8bf*)(ap + 8));
      const __bf16* bp = &UbS[w][nl][k0 + 16 * half];
      v16bf b = join16(*(const v8bf*)bp, *(const v8bf*)(bp + 8));
      acc = wmma_bf16(a, b, acc);
    }

    // LayerNorm partials: per-row sum/sumsq over this tile's 16 columns
#pragma unroll
    for (int r = 0; r < 8; ++r) {
      float v = acc[r], q = v * v;
#pragma unroll
      for (int msk = 1; msk < 16; msk <<= 1) {
        v += __shfl_xor(v, msk, 32);
        q += __shfl_xor(q, msk, 32);
      }
      if (nl == 0) {
        int row = r + 8 * half;
        atomicAdd(&rS1[row], v);
        atomicAdd(&rS2[row], q);
      }
    }
    __syncthreads();
    if (tid < 16)      atomicAdd(&lnstat[(t & 1) * 32 + tid], rS1[tid]);
    else if (tid < 32) atomicAdd(&lnstat[(t & 1) * 32 + tid], rS2[tid - 16]);

    gbarrier(bar, barIdx++, tid);

    if (tid < 16)      stS1[tid]      = lnstat[(t & 1) * 32 + tid];
    else if (tid < 32) stS2[tid - 16] = lnstat[(t & 1) * 32 + tid];
    __syncthreads();

    float gate[8];
#pragma unroll
    for (int r = 0; r < 8; ++r) {
      int m = r + 8 * half;
      float mu  = stS1[m] * (1.f / (float)GDIM);
      float var = stS2[m] * (1.f / (float)GDIM) - mu * mu;
      float un  = (acc[r] - mu) * rsqrtf(var + 1e-5f) * lng + lnb;
      float wxv = wx[((long)m * TLEN + t) * GDIM + col];
      gate[r] = wxv * bsc + bsh + un;
    }

    if (w >= 2) {
#pragma unroll
      for (int r = 0; r < 8; ++r) {
        int m = r + 8 * half;
        zS[m][(w - 2) * 16 + nl] = 1.f / (1.f + __expf(-gate[r]));
      }
    }
    __syncthreads();
    if (w < 2) {
#pragma unroll
      for (int r = 0; r < 8; ++r) {
        int m = r + 8 * half;
        float z  = zS[m][w * 16 + nl];
        float a  = fmaxf(gate[r], 0.f);
        float hn = z * hreg[r] + (1.f - z) * a;
        hreg[r] = hn;
        __bf16 hb = (__bf16)hn;
        h_g[m * HDIM + hcol] = hb;
        hbuf[((long)m * TLEN + t) * HDIM + hcol] = hb;
      }
    }
    gbarrier(bar, barIdx++, tid);
  }
}

// ---------------------------------------------------------------------------
// Host launcher
// ---------------------------------------------------------------------------
extern "C" void kernel_launch(void* const* d_in, const int* in_sizes, int n_in,
                              void* d_out, int out_size, void* d_ws, size_t ws_size,
                              hipStream_t stream) {
  (void)in_sizes; (void)n_in; (void)out_size;
  if (ws_size < (size_t)WS_NEED) return;

  const float* x    = (const float*)d_in[0];
  const float* W0   = (const float*)d_in[1];
  const float* U0   = (const float*)d_in[2];
  const float* bng0 = (const float*)d_in[3];
  const float* bnb0 = (const float*)d_in[4];
  const float* lng0 = (const float*)d_in[5];
  const float* lnb0 = (const float*)d_in[6];
  const float* W1   = (const float*)d_in[7];
  const float* U1   = (const float*)d_in[8];
  const float* bng1 = (const float*)d_in[9];
  const float* bnb1 = (const float*)d_in[10];
  const float* lng1 = (const float*)d_in[11];
  const float* lnb1 = (const float*)d_in[12];
  const float* Wd   = (const float*)d_in[13];
  const float* bd   = (const float*)d_in[14];

  char* ws = (char*)d_ws;
  __bf16* xb   = (__bf16*)(ws + OFF_XB);
  float*  wx   = (float*)(ws + OFF_WX);
  __bf16* hbuf = (__bf16*)(ws + OFF_HBUF);
  __bf16* W0b  = (__bf16*)(ws + OFF_W0B);
  __bf16* W1b  = (__bf16*)(ws + OFF_W1B);
  __bf16* Wdb  = (__bf16*)(ws + OFF_WDB);
  __bf16* U0b  = (__bf16*)(ws + OFF_U0B);
  __bf16* U1b  = (__bf16*)(ws + OFF_U1B);
  float*  bnsum   = (float*)(ws + OFF_BN);
  float*  bnsqs   = bnsum + GDIM;
  float*  bnscale = bnsum + 2 * GDIM;
  float*  bnshift = bnsum + 3 * GDIM;
  float*  ln0 = (float*)(ws + OFF_LN);
  float*  ln1 = ln0 + 64;
  __bf16* hg0 = (__bf16*)(ws + OFF_HG);
  __bf16* hg1 = hg0 + BATCH * HDIM;
  int*    bar0 = (int*)(ws + OFF_BAR);
  int*    bar1 = bar0 + 16;

  zero_kernel<<<64, 256, 0, stream>>>((unsigned*)(ws + OFF_BN), ZERO_U32);

  convert_pad_kernel<<<2048, 256, 0, stream>>>(x,  MROWS, FIN,  xb,  MROWS, KX);
  convert_pad_kernel<<<128, 256, 0, stream>>>(W0, GDIM,  FIN,  W0b, GDIM,  KX);
  convert_pad_kernel<<<256, 256, 0, stream>>>(W1, GDIM,  HDIM, W1b, GDIM,  HDIM);
  convert_pad_kernel<<<256, 256, 0, stream>>>(Wd, 1000,  HDIM, Wdb, GDIM,  HDIM);
  convert_pad_kernel<<<256, 256, 0, stream>>>(U0, GDIM,  HDIM, U0b, GDIM,  HDIM);
  convert_pad_kernel<<<256, 256, 0, stream>>>(U1, GDIM,  HDIM, U1b, GDIM,  HDIM);

  dim3 gwx(GDIM / 256, MROWS / 64);   // (4, 500)

  // Layer 0
  wx_gemm_kernel<<<gwx, 256, 0, stream>>>(xb, KX, KX, W0b, wx, bnsum, bnsqs);
  bn_finalize_kernel<<<4, 256, 0, stream>>>(bnsum, bnsqs, bng0, bnb0, bnscale, bnshift);
  sligru_recur_kernel<<<NWG, 128, 0, stream>>>(U0b, wx, bnscale, bnshift, lng0, lnb0,
                                               hbuf, hg0, ln0, bar0);
  // Layer 1
  wx_gemm_kernel<<<gwx, 256, 0, stream>>>(hbuf, HDIM, HDIM, W1b, wx, bnsum, bnsqs);
  bn_finalize_kernel<<<4, 256, 0, stream>>>(bnsum, bnsqs, bng1, bnb1, bnscale, bnshift);
  sligru_recur_kernel<<<NWG, 128, 0, stream>>>(U1b, wx, bnscale, bnshift, lng1, lnb1,
                                               hbuf, hg1, ln1, bar1);
  // Output head
  head_gemm_kernel<<<gwx, 256, 0, stream>>>(hbuf, Wdb, bd, (float*)d_out);
}
